// SPVCNN_24266565222405
// MI455X (gfx1250) — matrix operand order repaired and checked by reading, compiled, not run on hardware
//
#include <hip/hip_runtime.h>

// SPVCNN forward for MI455X (gfx1250, wave32).
// Heavy convs run as gather->WMMA GEMMs on v_wmma_f32_16x16x32_f16
// (f16 in, f32 accumulate). Each wave register-blocks a 32x32 output tile
// (4 accumulators -> 4 independent WMMAs per staged fragment set).
// Activations/weights pre-converted to f16 once per layer; everything is
// L2-resident (192 MB), so the limiter is WMMA issue + on-chip staging.

typedef __attribute__((ext_vector_type(16))) _Float16 v16h;
typedef __attribute__((ext_vector_type(8)))  _Float16 v8h;
typedef __attribute__((ext_vector_type(8)))  float    v8f;

#define BN_EPS 1e-5f
#define LRELU_SLOPE 0.01f

// ---------------------------------------------------------------- convert
__global__ void cvt_f16_kernel(const float* __restrict__ x,
                               _Float16* __restrict__ y, long n) {
  long i = (long)blockIdx.x * blockDim.x + threadIdx.x;
  if (i < n) y[i] = (_Float16)x[i];
}

// ---------------------------------------------------------------- WMMA sparse conv
// Y[M,Co] = sum_k sum_ci gather(Xh, nbr)[m,k,ci] * Wh[k,ci,co]
// nbr == nullptr -> identity gather (dense 1x1 conv, K must be 1).
// blockDim.x = 32 * (Co/32); Ci, Co multiples of 32. Block tile: 32 rows.
__global__ void sconv_wmma_kernel(const _Float16* __restrict__ Xh,
                                  const int* __restrict__ nbr,
                                  const _Float16* __restrict__ Wh,
                                  float* __restrict__ Y,
                                  int M, int K, int Ci, int Co) {
  __shared__ _Float16 As[32][40];   // 32 rows x 32 ch; row stride 80B (16B mult)
  __shared__ _Float16 Bs[32][136];  // 32 k x up-to-128 co; row stride 272B
  __shared__ int nbs[32];

  const int tid  = threadIdx.x;
  const int nt   = blockDim.x;
  const int wave = tid >> 5;
  const int lane = tid & 31;
  const int hf   = lane >> 4;
  const int l    = lane & 15;
  const int m0   = blockIdx.x * 32;
  const int co0  = wave * 32;

  v8f acc00 = {0.f, 0.f, 0.f, 0.f, 0.f, 0.f, 0.f, 0.f};
  v8f acc01 = acc00, acc10 = acc00, acc11 = acc00;

  for (int k = 0; k < K; ++k) {
    __syncthreads();
    if (tid < 32) {
      int m = m0 + tid;
      int nb = -1;
      if (m < M) {
        nb = nbr ? nbr[(long)m * K + k] : m;
        // software-prefetch next tap's gather row (global_prefetch_b8):
        // the neighbor indirection defeats any hardware prefetcher.
        if (nbr && (k + 1) < K) {
          int nb2 = nbr[(long)m * K + k + 1];
          if (nb2 >= 0) __builtin_prefetch(Xh + (long)nb2 * Ci, 0, 0);
        }
      }
      nbs[tid] = nb;
    }
    __syncthreads();

    for (int c0 = 0; c0 < Ci; c0 += 32) {
      // stage A: 32 rows x 32 ch, 128-bit vector loads, zeros for missing
      for (int i = tid; i < 128; i += nt) {
        int r = i >> 2, c = (i & 3) << 3;
        int nb = nbs[r];
        v8h v = {};
        if (nb >= 0) v = *(const v8h*)(Xh + (long)nb * Ci + c0 + c);
        *(v8h*)(&As[r][c]) = v;
      }
      // stage B: Wh[k, c0:c0+32, 0:Co] is contiguous -> coalesced b128 loads
      const _Float16* wp = Wh + ((long)k * Ci + c0) * Co;
      int nB = (32 * Co) >> 3;
      for (int i = tid; i < nB; i += nt) {
        int o = i << 3;
        int r = o / Co, c = o % Co;  // Co multiple of 8
        *(v8h*)(&Bs[r][c]) = *(const v8h*)(wp + o);
      }
      __syncthreads();

      // fragments: A and B both use contiguous k = hf*16 + e
      v16h a0, a1, b0, b1;
#pragma unroll
      for (int e = 0; e < 16; ++e) {
        int j  = e >> 1;
        int kk = hf * 16 + e;
        a0[e] = As[l][kk];
        a1[e] = As[16 + l][kk];
        b0[e] = Bs[hf * 16 + e][co0 + l];
        b1[e] = Bs[hf * 16 + e][co0 + 16 + l];
      }
      // 4 independent WMMAs: no inter-op RAW -> no hazard NOPs
      acc00 = __builtin_amdgcn_wmma_f32_16x16x32_f16(
          false, a0, false, b0, (short)0, acc00, false, false);
      acc01 = __builtin_amdgcn_wmma_f32_16x16x32_f16(
          false, a0, false, b1, (short)0, acc01, false, false);
      acc10 = __builtin_amdgcn_wmma_f32_16x16x32_f16(
          false, a1, false, b0, (short)0, acc10, false, false);
      acc11 = __builtin_amdgcn_wmma_f32_16x16x32_f16(
          false, a1, false, b1, (short)0, acc11, false, false);
      __syncthreads();
    }
  }

  // D layout: VGPR r -> row (r + 8*half), col = lane%16
#pragma unroll
  for (int r = 0; r < 8; ++r) {
    int mA = m0 + r + hf * 8;
    int mB = mA + 16;
    if (mA < M) {
      Y[(long)mA * Co + co0 + l]      = acc00[r];
      Y[(long)mA * Co + co0 + 16 + l] = acc01[r];
    }
    if (mB < M) {
      Y[(long)mB * Co + co0 + l]      = acc10[r];
      Y[(long)mB * Co + co0 + 16 + l] = acc11[r];
    }
  }
}

// ---------------------------------------------------------------- BN stats
// stats[0..C) = mean, stats[C..2C) = var (population), one block per channel.
__global__ void bn_stats_kernel(const float* __restrict__ X,
                                float* __restrict__ stats, int M, int C) {
  __shared__ float sh[256], sh2[256];
  int c = blockIdx.x;
  float s = 0.f, s2 = 0.f;
  for (int m = threadIdx.x; m < M; m += blockDim.x) {
    float v = X[(long)m * C + c];
    s += v;
    s2 += v * v;
  }
  sh[threadIdx.x] = s;
  sh2[threadIdx.x] = s2;
  __syncthreads();
  for (int o = blockDim.x >> 1; o > 0; o >>= 1) {
    if (threadIdx.x < o) {
      sh[threadIdx.x]  += sh[threadIdx.x + o];
      sh2[threadIdx.x] += sh2[threadIdx.x + o];
    }
    __syncthreads();
  }
  if (threadIdx.x == 0) {
    float mean = sh[0] / (float)M;
    stats[c]     = mean;
    stats[C + c] = sh2[0] / (float)M - mean * mean;
  }
}

// mode: 0 none, 1 relu, 2 leaky-relu. add (optional) applied after BN.
__global__ void bn_apply_kernel(const float* __restrict__ X,
                                const float* __restrict__ stats,
                                const float* __restrict__ g,
                                const float* __restrict__ b,
                                const float* __restrict__ add,
                                float* __restrict__ Y,
                                long n, int C, int mode) {
  long i = (long)blockIdx.x * blockDim.x + threadIdx.x;
  if (i >= n) return;
  int c = (int)(i % C);
  float v = (X[i] - stats[c]) * rsqrtf(stats[C + c] + BN_EPS) * g[c] + b[c];
  if (add) v += add[i];
  if (mode == 1) v = fmaxf(v, 0.f);
  else if (mode == 2) v = (v >= 0.f) ? v : LRELU_SLOPE * v;
  Y[i] = v;
}

// ---------------------------------------------------------------- transposed conv
// Y[m,co] = sum_ci X[par[m],ci] * W[kidx[m],ci,co]   (W: [8,Ci,Co])
__global__ void tconv_kernel(const float* __restrict__ X,
                             const int* __restrict__ par,
                             const int* __restrict__ kidx,
                             const float* __restrict__ W,
                             float* __restrict__ Y, int M, int Ci, int Co) {
  long i = (long)blockIdx.x * blockDim.x + threadIdx.x;
  if (i >= (long)M * Co) return;
  int m = (int)(i / Co), co = (int)(i % Co);
  const float* xr = X + (long)par[m] * Ci;
  const float* wr = W + (long)kidx[m] * Ci * Co + co;
  float acc = 0.f;
  for (int ci = 0; ci < Ci; ++ci) acc += xr[ci] * wr[(long)ci * Co];
  Y[i] = acc;
}

// ---------------------------------------------------------------- concat
__global__ void concat_kernel(const float* __restrict__ A, int Ca,
                              const float* __restrict__ B, int Cb,
                              float* __restrict__ Y, int M) {
  int C = Ca + Cb;
  long i = (long)blockIdx.x * blockDim.x + threadIdx.x;
  if (i >= (long)M * C) return;
  int m = (int)(i / C), c = (int)(i % C);
  Y[i] = (c < Ca) ? A[(long)m * Ca + c] : B[(long)m * Cb + (c - Ca)];
}

// ================================================================ host
extern "C" void kernel_launch(void* const* d_in, const int* in_sizes, int n_in,
                              void* d_out, int out_size, void* d_ws, size_t ws_size,
                              hipStream_t stream) {
  (void)n_in; (void)out_size; (void)ws_size;
  auto pf = [&](int i) { return (const float*)d_in[i]; };
  auto pi = [&](int i) { return (const int*)d_in[i]; };

  // Input order = setup_inputs() dict insertion order:
  // 0:feats, 1..75: params in construction order, 76..84: index maps.
  const int N0 = in_sizes[0] / 64;
  const int M1 = in_sizes[77] / 27;
  const int M2 = in_sizes[78] / 27;

  const int* nbr0   = pi(76);
  const int* nbr1   = pi(77);
  const int* nbr2   = pi(78);
  const int* down01 = pi(79);
  const int* down12 = pi(80);
  const int* par10  = pi(81);
  const int* kidx10 = pi(82);
  const int* par21  = pi(83);
  const int* kidx21 = pi(84);

  // workspace bump allocator
  char* base = (char*)d_ws;
  size_t off = 0;
  auto walloc = [&](size_t bytes) -> void* {
    void* p = base + off;
    off += (bytes + 255) & ~(size_t)255;
    return p;
  };

  size_t maxM12 = (size_t)((M1 > M2) ? M1 : M2);
  size_t genE = maxM12 * 128;                          // deepest 128-ch tensors
  if ((size_t)N0 * 64 > genE) genE = (size_t)N0 * 64;  // level-0 concat

  float* tA   = (float*)walloc(genE * 4);  // conv outputs
  float* tB   = (float*)walloc(genE * 4);  // layer i/o
  float* tC   = (float*)walloc(genE * 4);  // shortcut branch
  float* tD   = (float*)walloc(genE * 4);  // res-block hidden
  float* tCat = (float*)walloc(genE * 4);  // concat buffer
  float* x0   = (float*)walloc((size_t)N0 * 32 * 4);
  float* x1   = (float*)walloc((size_t)M1 * 64 * 4);
  float* x2   = (float*)walloc((size_t)M2 * 128 * 4);
  _Float16* Xh = (_Float16*)walloc(genE * 2);
  _Float16* Wh = (_Float16*)walloc((size_t)27 * 128 * 128 * 2);
  float* stats = (float*)walloc(256 * 4);

  auto cvt = [&](const float* x, _Float16* y, long n) {
    unsigned blocks = (unsigned)((n + 255) / 256);
    cvt_f16_kernel<<<dim3(blocks), dim3(256), 0, stream>>>(x, y, n);
  };
  auto conv = [&](const float* X, int Nin, const int* nbr, const float* W,
                  float* Y, int M, int K, int Ci, int Co) {
    cvt(X, Xh, (long)Nin * Ci);
    cvt(W, Wh, (long)K * Ci * Co);
    sconv_wmma_kernel<<<dim3((M + 31) / 32), dim3(32 * (Co / 32)), 0, stream>>>(
        Xh, nbr, Wh, Y, M, K, Ci, Co);
  };
  auto bn = [&](const float* X, const float* g, const float* b,
                const float* add, float* Y, int M, int C, int mode) {
    bn_stats_kernel<<<dim3(C), dim3(256), 0, stream>>>(X, stats, M, C);
    long n = (long)M * C;
    bn_apply_kernel<<<dim3((unsigned)((n + 255) / 256)), dim3(256), 0, stream>>>(
        X, stats, g, b, add, Y, n, C, mode);
  };
  auto tconv = [&](const float* X, const int* par, const int* kidx,
                   const float* W, float* Y, int M, int Ci, int Co) {
    long n = (long)M * Co;
    tconv_kernel<<<dim3((unsigned)((n + 255) / 256)), dim3(256), 0, stream>>>(
        X, par, kidx, W, Y, M, Ci, Co);
  };
  auto concat = [&](const float* A, int Ca, const float* B, int Cb,
                    float* Y, int M) {
    long n = (long)M * (Ca + Cb);
    concat_kernel<<<dim3((unsigned)((n + 255) / 256)), dim3(256), 0, stream>>>(
        A, Ca, B, Cb, Y, M);
  };
  // residual block: out = lrelu(bn(conv2(lrelu(bn(conv1(x))))) + shortcut)
  auto res = [&](const float* x, const int* nbr, int M, int Ci, int Co,
                 int pb, bool hasWd, float* out) {
    conv(x, M, nbr, pf(pb + 0), tA, M, 27, Ci, Co);
    bn(tA, pf(pb + 1), pf(pb + 2), nullptr, tD, M, Co, 2);
    conv(tD, M, nbr, pf(pb + 3), tA, M, 27, Co, Co);
    const float* sc = x;
    if (hasWd) {
      conv(x, M, nullptr, pf(pb + 6), tC, M, 1, Ci, Co);  // 1x1 via WMMA
      bn(tC, pf(pb + 7), pf(pb + 8), nullptr, tC, M, Co, 0);
      sc = tC;
    }
    bn(tA, pf(pb + 4), pf(pb + 5), sc, out, M, Co, 2);
  };

  // ---------------- forward pass ----------------
  // stem: relu(bn(sconv27(feats)))
  conv(pf(0), N0, nbr0, pf(1), tA, N0, 27, 64, 32);
  bn(tA, pf(2), pf(3), nullptr, x0, N0, 32, 1);
  // down 0->1
  conv(x0, N0, down01, pf(4), tA, M1, 8, 32, 32);
  bn(tA, pf(5), pf(6), nullptr, tB, M1, 32, 2);
  res(tB, nbr1, M1, 32, 64, 7, true, tB);    // s1r1
  res(tB, nbr1, M1, 64, 64, 16, false, x1);  // s1r2
  // down 1->2
  conv(x1, M1, down12, pf(22), tA, M2, 8, 64, 64);
  bn(tA, pf(23), pf(24), nullptr, tB, M2, 64, 2);
  res(tB, nbr2, M2, 64, 128, 25, true, tB);    // s2r1
  res(tB, nbr2, M2, 128, 128, 34, false, x2);  // s2r2
  // up 2->1
  tconv(x2, par21, kidx21, pf(40), tA, M1, 128, 64);
  bn(tA, pf(41), pf(42), nullptr, tB, M1, 64, 2);
  concat(tB, 64, x1, 64, tCat, M1);
  res(tCat, nbr1, M1, 128, 64, 43, true, tB);  // u1r1
  res(tB, nbr1, M1, 64, 64, 52, false, tB);    // u1r2
  // up 1->0
  tconv(tB, par10, kidx10, pf(58), tA, N0, 64, 32);
  bn(tA, pf(59), pf(60), nullptr, tB, N0, 32, 2);
  concat(tB, 32, x0, 32, tCat, N0);
  res(tCat, nbr0, N0, 64, 32, 61, true, tB);        // u2r1
  res(tB, nbr0, N0, 32, 32, 70, false, (float*)d_out);  // u2r2 -> output
}
